// FROStNET_57080115364024
// MI455X (gfx1250) — matrix operand order repaired and checked by reading, not compiled, not run on hardware
//
#include <hip/hip_runtime.h>
#include <hip/hip_bf16.h>

// ---------------- problem constants ----------------
#define NP     1024          // P
#define NB     16            // batch
#define NT     2049          // delay rows
#define TPAD   2064          // 129 * 16
#define NTT    129           // t tiles
#define SWN    3072          // 3*P

// ---------------- workspace layout -----------------
#define OFF_SW     0u                        // float[3072]
#define OFF_MAX    16384u                    // uint[16]
#define OFF_COSH   32768u                    // __bf16[1<<20] fragment-swizzled, hi
#define OFF_COSL   (OFF_COSH + (1u<<21))     // lo
#define OFF_SINH   (OFF_COSL + (1u<<21))     // hi
#define OFF_SINL   (OFF_SINH + (1u<<21))     // lo
#define OFF_TRACE  (16u<<20)                 // float[16*1024*2064] ~135MB (ws >= ~152MB)

typedef __attribute__((ext_vector_type(16))) __bf16 v16bf;
typedef __attribute__((ext_vector_type(8)))  float  v8f;

#define WMMA_BF16(A, Bm, C) \
    __builtin_amdgcn_wmma_f32_16x16x32_bf16(false, A, false, Bm, (short)0, C, false, false)

// ---- 1) resample switch 2048 -> 3072 (half-pixel linear, edge clamp); zero maxes ----
__global__ __launch_bounds__(256) void frog_init_sw(const float* __restrict__ swin,
                                                    float* __restrict__ sw,
                                                    unsigned* __restrict__ mx) {
    int tid = blockIdx.x * 256 + threadIdx.x;
    if (tid < SWN) {
        float c = (tid + 0.5f) * (2048.0f / 3072.0f) - 0.5f;
        float f = floorf(c);
        int   i0 = (int)f;
        float w  = c - f;
        int ia = min(max(i0,     0), 2047);
        int ib = min(max(i0 + 1, 0), 2047);
        sw[tid] = (1.0f - w) * swin[ia] + w * swin[ib];
    }
    if (tid < NB) mx[tid] = 0u;
}

// ---- 2) DFT cos/sin matrices as hi/lo bf16 pairs, pre-swizzled into B-fragment order ----
// id = kt<<14 | nc<<9 | lane<<4 | j : column kf = kt*16 + (lane&15),
// row n = nc*32 + (lane>=16 ? 16:0) + j.  fftshift folded: m=(kf+512)&1023.
__global__ __launch_bounds__(256) void frog_init_dft(__bf16* __restrict__ cosH,
                                                     __bf16* __restrict__ cosL,
                                                     __bf16* __restrict__ sinH,
                                                     __bf16* __restrict__ sinL) {
    unsigned id = blockIdx.x * 256 + threadIdx.x;        // 1<<20 total
    int j    =  id        & 15;
    int lane = (id >> 4)  & 31;
    int nc   = (id >> 9)  & 31;
    int kt   =  id >> 14;
    int kf = kt * 16 + (lane & 15);
    int n  = nc * 32 + ((lane >> 4) << 4) + j;
    int m  = (kf + 512) & 1023;
    int ph = (m * n) & 1023;                              // exact mod-N phase
    float ang = (float)ph * (6.28318530717958647692f / 1024.0f);
    float s, c;
    __sincosf(ang, &s, &c);
    __bf16 ch = (__bf16)c;
    __bf16 sh = (__bf16)s;
    cosH[id] = ch;  cosL[id] = (__bf16)(c - (float)ch);
    sinH[id] = sh;  sinL[id] = (__bf16)(s - (float)sh);
}

// ---- 3) fused window * split-bf16 DFT-GEMM -> |.|^2 -> trace + per-batch max ----
// grid: (4 kf-groups, 129 t-tiles, 16 batches), block 256 = 8 wave32 waves.
// Each wave owns TWO 16x16 (t x kf) tiles; 12 WMMAs per K-chunk per wave
// (3-term bf16 split: AhBh + AlBh + AhBl, Re & Im, 2 tiles).
__global__ __launch_bounds__(256) void frog_gemm(const float* __restrict__ pulse,
                                                 const float* __restrict__ sw,
                                                 const __bf16* __restrict__ cosH,
                                                 const __bf16* __restrict__ cosL,
                                                 const __bf16* __restrict__ sinH,
                                                 const __bf16* __restrict__ sinL,
                                                 float* __restrict__ trace,
                                                 unsigned* __restrict__ mx) {
    __shared__ float s_sw[SWN + 32];   // pad: garbage rows of last t-tile stay in-bounds
    __shared__ float s_pu[NP];

    const int b   = blockIdx.z;
    const int tt  = blockIdx.y;
    const int kfg = blockIdx.x;
    const int tid = threadIdx.x;

    for (int i = tid; i < SWN + 32; i += 256) s_sw[i] = (i < SWN) ? sw[i] : 0.0f;
    for (int i = tid; i < NP;       i += 256) s_pu[i] = pulse[b * NP + i];
    __syncthreads();

    const int wave  = tid >> 5;
    const int lane  = tid & 31;
    const int kt0   = (kfg * 8 + wave) * 2;    // first of two 16-col tiles (0..62)
    const int khalf = lane >> 4;
    const int t0    = tt * 16;
    const int trow  = t0 + (lane & 15);

    v8f accR0 = {}, accI0 = {}, accR1 = {}, accI1 = {};

#pragma unroll 2
    for (int nc = 0; nc < 32; ++nc) {
        // A fragment (16x32) split into hi/lo bf16, contiguous k = 16*khalf + j (same order as B)
        v16bf ah, al;
#pragma unroll
        for (int j = 0; j < 16; ++j) {
            int k = j + (khalf << 4);   // K index 0..31
            int n = (nc << 5) + k;
            float v = s_sw[trow + n] * s_pu[n];
            __bf16 h = (__bf16)v;
            ah[j] = h;
            al[j] = (__bf16)(v - (float)h);
        }
        // B fragments: 32B contiguous per lane thanks to pre-swizzle
        size_t f0 = (((size_t)(kt0 * 32 + nc)) * 32 + lane) * 16;
        size_t f1 = f0 + (size_t)32 * 32 * 16;              // kt0+1 plane
        v16bf ch0 = *(const v16bf*)(cosH + f0);
        v16bf cl0 = *(const v16bf*)(cosL + f0);
        v16bf sh0 = *(const v16bf*)(sinH + f0);
        v16bf sl0 = *(const v16bf*)(sinL + f0);
        v16bf ch1 = *(const v16bf*)(cosH + f1);
        v16bf cl1 = *(const v16bf*)(cosL + f1);
        v16bf sh1 = *(const v16bf*)(sinH + f1);
        v16bf sl1 = *(const v16bf*)(sinL + f1);

        // interleave 4 independent accumulator chains for ILP
        accR0 = WMMA_BF16(ah, ch0, accR0);
        accI0 = WMMA_BF16(ah, sh0, accI0);
        accR1 = WMMA_BF16(ah, ch1, accR1);
        accI1 = WMMA_BF16(ah, sh1, accI1);
        accR0 = WMMA_BF16(al, ch0, accR0);
        accI0 = WMMA_BF16(al, sh0, accI0);
        accR1 = WMMA_BF16(al, ch1, accR1);
        accI1 = WMMA_BF16(al, sh1, accI1);
        accR0 = WMMA_BF16(ah, cl0, accR0);
        accI0 = WMMA_BF16(ah, sl0, accI0);
        accR1 = WMMA_BF16(ah, cl1, accR1);
        accI1 = WMMA_BF16(ah, sl1, accI1);
    }

    // power; only the last t-tile has invalid (padding) rows -> uniform branch
    const int tb = t0 + (khalf << 3);
    float p0[8], p1[8];
    float m = 0.0f;
#pragma unroll
    for (int i = 0; i < 8; ++i) {
        p0[i] = accR0[i] * accR0[i] + accI0[i] * accI0[i];
        p1[i] = accR1[i] * accR1[i] + accI1[i] * accI1[i];
    }
    if (tt != NTT - 1) {
#pragma unroll
        for (int i = 0; i < 8; ++i) m = fmaxf(m, fmaxf(p0[i], p1[i]));
    } else {
#pragma unroll
        for (int i = 0; i < 8; ++i)
            if (tb + i <= NT - 1) m = fmaxf(m, fmaxf(p0[i], p1[i]));
    }

    // store 8 contiguous t values per tile: trace[b][kf][tb..tb+7]
    const int kf0 = kt0 * 16 + (lane & 15);
    float* tp0 = trace + ((size_t)(b * NP + kf0)) * TPAD + tb;
    float* tp1 = tp0 + (size_t)16 * TPAD;                   // kf0 + 16
    ((float4*)tp0)[0] = make_float4(p0[0], p0[1], p0[2], p0[3]);
    ((float4*)tp0)[1] = make_float4(p0[4], p0[5], p0[6], p0[7]);
    ((float4*)tp1)[0] = make_float4(p1[0], p1[1], p1[2], p1[3]);
    ((float4*)tp1)[1] = make_float4(p1[4], p1[5], p1[6], p1[7]);

    // wave32 max reduction + atomic max (values >= 0 -> uint compare is order-safe)
    for (int off = 16; off; off >>= 1)
        m = fmaxf(m, __shfl_xor(m, off, 32));
    if (lane == 0)
        atomicMax(mx + b, __float_as_uint(m));
}

// ---- 4) resize t: 2049 -> 1024 (half-pixel linear) + normalize ----
__global__ __launch_bounds__(256) void frog_resize(const float* __restrict__ trace,
                                                   const unsigned* __restrict__ mx,
                                                   float* __restrict__ out) {
    unsigned gid = blockIdx.x * 256 + threadIdx.x;    // 1<<24 total
    int j  =  gid        & 1023;
    int kf = (gid >> 10) & 1023;
    int b  =  gid >> 20;
    float c = (j + 0.5f) * (2049.0f / 1024.0f) - 0.5f;
    float f = floorf(c);
    int   ti = min(max((int)f, 0), NT - 2);
    float w  = c - f;
    const float* tp = trace + ((size_t)(b * NP + kf)) * TPAD + ti;
    float v0 = tp[0], v1 = tp[1];
    float v  = v0 + w * (v1 - v0);
    out[gid] = v / __uint_as_float(mx[b]);
}

extern "C" void kernel_launch(void* const* d_in, const int* in_sizes, int n_in,
                              void* d_out, int out_size, void* d_ws, size_t ws_size,
                              hipStream_t stream) {
    const float* pulse = (const float*)d_in[0];   // (16, 1024) f32
    const float* swin  = (const float*)d_in[1];   // (1, 2048)  f32
    // d_in[2] = points (always 1024)
    float*    out   = (float*)d_out;              // (16,1024,1024,1) f32
    char*     ws    = (char*)d_ws;                // assumes ws >= ~152 MB
    float*    sw    = (float*)(ws + OFF_SW);
    unsigned* mx    = (unsigned*)(ws + OFF_MAX);
    __bf16*   cosH  = (__bf16*)(ws + OFF_COSH);
    __bf16*   cosL  = (__bf16*)(ws + OFF_COSL);
    __bf16*   sinH  = (__bf16*)(ws + OFF_SINH);
    __bf16*   sinL  = (__bf16*)(ws + OFF_SINL);
    float*    trace = (float*)(ws + OFF_TRACE);

    frog_init_sw <<<12,    256, 0, stream>>>(swin, sw, mx);
    frog_init_dft<<<4096,  256, 0, stream>>>(cosH, cosL, sinH, sinL);
    frog_gemm    <<<dim3(4, NTT, NB), 256, 0, stream>>>(pulse, sw, cosH, cosL, sinH, sinL,
                                                        trace, mx);
    frog_resize  <<<65536, 256, 0, stream>>>(trace, mx, out);
}
